// Decoder_69552700391978
// MI455X (gfx1250) — matrix operand order repaired and checked by reading, not compiled, not run on hardware
//
#include <hip/hip_runtime.h>
#include <hip/hip_bf16.h>
#include <math.h>

#define HH   512
#define WDD  256
#define VV   32000
#define BB   16
#define TT   64
#define SS   256
#define KK   32
#define H3   1536
#define DIN  768
#define SWIN 2304
#define NEGF (-1e30f)

typedef __attribute__((ext_vector_type(16))) __bf16 v16bf;
typedef __attribute__((ext_vector_type(8)))  float  v8f;

__device__ __forceinline__ float sigm(float x) { return 1.0f / (1.0f + expf(-x)); }

__device__ __forceinline__ v8f wmma_bf16(v16bf a, v16bf b, v8f c) {
    // D = A(16x32) * B(32x16) + C(16x16), f32 accumulate
    return __builtin_amdgcn_wmma_f32_16x16x32_bf16(false, a, false, b, (short)0, c,
                                                   false, false);
}

// ---------------------------------------------------------------------------
// Generic C = A * B^T (+bias) (+add), A:(M,K) lda, Bw:(N,K) ldb, C:(M,N) ldc.
// One wave computes MREP consecutive 16x16 M-tiles for one 16-wide N tile,
// reusing the B fragment MREP times.
// ---------------------------------------------------------------------------
template <int MREP>
__global__ void gemm_bt(const float* __restrict__ A, int lda,
                        const float* __restrict__ Bw, int ldb,
                        const float* __restrict__ bias,
                        const float* __restrict__ add, int ldadd,
                        float* __restrict__ C, int ldc, int M, int N, int K) {
    const int lane = threadIdx.x & 31;
    const int wave = blockIdx.x * (blockDim.x >> 5) + (threadIdx.x >> 5);
    const int ntiles = N >> 4;
    const int nstrips = (M + 16 * MREP - 1) / (16 * MREP);
    if (wave >= ntiles * nstrips) return;
    const int nt = wave % ntiles;
    const int ms = wave / ntiles;
    const int n0 = nt << 4;
    const int m0 = ms * (16 * MREP);

    v8f acc[MREP] = {};
    const int bn  = n0 + (lane & 15);       // B: lane owns column n, K-range by half
    const int bko = (lane >> 4) << 4;       // 0 or 16
    const int am  = lane & 15;              // A: lane owns row m
    const int ako = (lane >> 4) << 4;       // 0 or 16

    for (int k = 0; k < K; k += 32) {
        v16bf bfrag;
        const float* bp = Bw + (size_t)bn * ldb + (k + bko);
#pragma unroll
        for (int i = 0; i < 16; i++) bfrag[i] = (__bf16)bp[i];
#pragma unroll
        for (int r = 0; r < MREP; r++) {
            const int m = m0 + (r << 4) + am;
            const float* ap = A + (size_t)m * lda + k;
            v16bf afrag;
#pragma unroll
            for (int j = 0; j < 8; j++) {
                const int kk = ako + (j << 1);
                afrag[2 * j]     = (__bf16)ap[kk];
                afrag[2 * j + 1] = (__bf16)ap[kk + 1];
            }
            acc[r] = wmma_bf16(afrag, bfrag, acc[r]);
        }
    }
    const int col = lane & 15;
    const int rbase = (lane >> 4) << 3;
    const float bv = bias ? bias[n0 + col] : 0.0f;
#pragma unroll
    for (int r = 0; r < MREP; r++) {
#pragma unroll
        for (int rr = 0; rr < 8; rr++) {
            const int m = m0 + (r << 4) + rbase + rr;
            float v = acc[r][rr] + bv;
            if (add) v += add[(size_t)m * ldadd + n0 + col];
            C[(size_t)m * ldc + n0 + col] = v;
        }
    }
}

// Variant with pre-converted bf16 B matrix (kept L2-resident for the vocab GEMM).
template <int MREP>
__global__ void gemm_btb(const float* __restrict__ A, int lda,
                         const unsigned short* __restrict__ Bw16, int ldb,
                         const float* __restrict__ bias,
                         float* __restrict__ C, int ldc, int M, int N, int K) {
    const int lane = threadIdx.x & 31;
    const int wave = blockIdx.x * (blockDim.x >> 5) + (threadIdx.x >> 5);
    const int ntiles = N >> 4;
    const int nstrips = (M + 16 * MREP - 1) / (16 * MREP);
    if (wave >= ntiles * nstrips) return;
    const int nt = wave % ntiles;
    const int ms = wave / ntiles;
    const int n0 = nt << 4;
    const int m0 = ms * (16 * MREP);

    v8f acc[MREP] = {};
    const int bn  = n0 + (lane & 15);
    const int bko = (lane >> 4) << 4;
    const int am  = lane & 15;
    const int ako = (lane >> 4) << 4;
    const __bf16* Bb = (const __bf16*)Bw16;

    for (int k = 0; k < K; k += 32) {
        v16bf bfrag;
        const __bf16* bp = Bb + (size_t)bn * ldb + (k + bko);
#pragma unroll
        for (int i = 0; i < 16; i++) bfrag[i] = bp[i];
#pragma unroll
        for (int r = 0; r < MREP; r++) {
            const int m = m0 + (r << 4) + am;
            const float* ap = A + (size_t)m * lda + k;
            v16bf afrag;
#pragma unroll
            for (int j = 0; j < 8; j++) {
                const int kk = ako + (j << 1);
                afrag[2 * j]     = (__bf16)ap[kk];
                afrag[2 * j + 1] = (__bf16)ap[kk + 1];
            }
            acc[r] = wmma_bf16(afrag, bfrag, acc[r]);
        }
    }
    const int col = lane & 15;
    const int rbase = (lane >> 4) << 3;
    const float bv = bias ? bias[n0 + col] : 0.0f;
#pragma unroll
    for (int r = 0; r < MREP; r++) {
#pragma unroll
        for (int rr = 0; rr < 8; rr++) {
            const int m = m0 + (r << 4) + rbase + rr;
            C[(size_t)m * ldc + n0 + col] = acc[r][rr] + bv;
        }
    }
}

__global__ void cvt_bf16(const float* __restrict__ in, unsigned short* __restrict__ out,
                         size_t n) {
    size_t i = ((size_t)blockIdx.x * blockDim.x + threadIdx.x) * 8;
    if (i >= n) return;
    __bf16* o = (__bf16*)out;
#pragma unroll
    for (int j = 0; j < 8; j++) o[i + j] = (__bf16)in[i + j];
}

// ---------------------------------------------------------------------------
// dec_in[b,t,:] = concat(cembeds[b], (t==0 ? 0 : emb[r_utts[b,t-1]]))
// ---------------------------------------------------------------------------
__global__ void build_decin(const float* __restrict__ cembeds,
                            const float* __restrict__ emb,
                            const int* __restrict__ r_utts,
                            float* __restrict__ dec_in) {
    int idx = blockIdx.x * blockDim.x + threadIdx.x;
    if (idx >= BB * TT * DIN) return;
    int c = idx % DIN;
    int bt = idx / DIN;
    int t = bt % TT, b = bt / TT;
    float v;
    if (c < HH) v = cembeds[b * HH + c];
    else {
        int wd = c - HH;
        v = (t == 0) ? 0.0f : emb[(size_t)r_utts[b * TT + t - 1] * WDD + wd];
    }
    dec_in[idx] = v;
}

// ---------------------------------------------------------------------------
// Fused gh = h @ Whh^T + bhh and GRU elementwise.
// ---------------------------------------------------------------------------
__global__ void gru_step(const float* __restrict__ GIt, int gi_stride,
                         const float* __restrict__ h,
                         const float* __restrict__ Whh,
                         const float* __restrict__ bhh,
                         const int* __restrict__ r_ulens, int t,
                         float* __restrict__ hs3, float* __restrict__ hnext) {
    const int lane = threadIdx.x & 31;
    const int wave = blockIdx.x * (blockDim.x >> 5) + (threadIdx.x >> 5);
    if (wave >= HH / 16) return;
    const int n0 = wave << 4;
    v8f ar = {}, az = {}, an = {};
    const int am  = lane & 15;
    const int ako = (lane >> 4) << 4;
    const int bn  = n0 + (lane & 15);
    const int bko = (lane >> 4) << 4;
    for (int k = 0; k < HH; k += 32) {
        v16bf afrag;
        const float* ap = h + am * HH + k;
#pragma unroll
        for (int j = 0; j < 8; j++) {
            const int kk = ako + (j << 1);
            afrag[2 * j]     = (__bf16)ap[kk];
            afrag[2 * j + 1] = (__bf16)ap[kk + 1];
        }
        v16bf br, bz, bn2;
        const float* bpr = Whh + (size_t)bn * HH + k + bko;
        const float* bpz = Whh + (size_t)(bn + 512) * HH + k + bko;
        const float* bpn = Whh + (size_t)(bn + 1024) * HH + k + bko;
#pragma unroll
        for (int i = 0; i < 16; i++) {
            br[i]  = (__bf16)bpr[i];
            bz[i]  = (__bf16)bpz[i];
            bn2[i] = (__bf16)bpn[i];
        }
        ar = wmma_bf16(afrag, br, ar);
        az = wmma_bf16(afrag, bz, az);
        an = wmma_bf16(afrag, bn2, an);
    }
    const int col = lane & 15;
    const int rbase = (lane >> 4) << 3;
    const int o = n0 + col;
#pragma unroll
    for (int rr = 0; rr < 8; rr++) {
        const int m = rbase + rr;  // batch row (0..15)
        float ir  = GIt[(size_t)m * gi_stride + o];
        float iz  = GIt[(size_t)m * gi_stride + 512 + o];
        float inn = GIt[(size_t)m * gi_stride + 1024 + o];
        float hr = ar[rr] + bhh[o];
        float hz = az[rr] + bhh[512 + o];
        float hn = an[rr] + bhh[1024 + o];
        float r = sigm(ir + hr);
        float z = sigm(iz + hz);
        float hv = h[m * HH + o];
        float hnew = (1.0f - z) * tanhf(inn + r * hn) + z * hv;
        hs3[m * H3 + o] = hnew;
        hnext[m * HH + o] = (t < r_ulens[m]) ? hnew : hv;
    }
}

// ---------------------------------------------------------------------------
// Context attention.  Logit phase: one wave per s-row, lanes stride the
// 512-wide reduction (coalesced), reduce via shuffles.  Then masked softmax
// and h_att = attn @ c_uh.
// ---------------------------------------------------------------------------
__global__ void attn_step(const float* __restrict__ HA, const float* __restrict__ ST,
                          const float* __restrict__ Wvw, const float* __restrict__ Wvb,
                          const int* __restrict__ c_utts, const float* __restrict__ c_uh,
                          float* __restrict__ hs3, float* __restrict__ attn_all, int t) {
    __shared__ float stl[HH];
    __shared__ float wvl[HH];
    __shared__ float lg[SS];
    __shared__ float red[SS];
    __shared__ float sh[SS];
    __shared__ int anyv;
    const int b = blockIdx.x, tid = threadIdx.x;
    const int lane = tid & 31, wave = tid >> 5;
    for (int i = tid; i < HH; i += 256) {
        stl[i] = ST[b * HH + i];
        wvl[i] = Wvw[i];
    }
    if (tid == 0) anyv = 0;
    __syncthreads();
    for (int j = 0; j < 32; j++) {
        const int s = wave * 32 + j;
        const float* hap = HA + ((size_t)(b * SS + s)) * HH;
        float part = 0.0f;
        for (int i = lane; i < HH; i += 32) {
            float v = hap[i] + stl[i];
            part += fmaxf(v, 0.0f) * wvl[i];
        }
        for (int off = 16; off > 0; off >>= 1) part += __shfl_xor(part, off, 32);
        if (lane == 0) lg[s] = part + Wvb[0];
    }
    __syncthreads();
    const int s = tid;
    const int notm = (c_utts[b * SS + s] != 0);
    float l = notm ? lg[s] : NEGF;
    if (notm) atomicOr(&anyv, 1);
    red[s] = l;
    __syncthreads();
    for (int off = 128; off > 0; off >>= 1) {
        if (s < off) red[s] = fmaxf(red[s], red[s + off]);
        __syncthreads();
    }
    float mx = red[0];
    __syncthreads();
    float e = expf(l - mx);
    red[s] = e;
    __syncthreads();
    for (int off = 128; off > 0; off >>= 1) {
        if (s < off) red[s] += red[s + off];
        __syncthreads();
    }
    float sum = red[0];
    float a = anyv ? (e / sum) : 0.0f;
    sh[s] = a;
    attn_all[((size_t)b * TT + t) * SS + s] = a;
    __syncthreads();
    float a0 = 0.0f, a1 = 0.0f;
    for (int sp = 0; sp < SS; sp++) {
        float av = sh[sp];
        const float* cp = c_uh + ((size_t)(b * SS + sp)) * HH;
        a0 += av * cp[s];
        a1 += av * cp[s + 256];
    }
    hs3[b * H3 + 512 + s] = a0;
    hs3[b * H3 + 512 + s + 256] = a1;
}

// ---------------------------------------------------------------------------
// Key attention: klog = relu(gk + hk).k2w + k2b, masked softmax over K=32,
// h_key = key_attn @ gembeds.
// ---------------------------------------------------------------------------
__global__ void key_step(const float* __restrict__ GK, const float* __restrict__ HK,
                         const float* __restrict__ k2w, const float* __restrict__ k2b,
                         const int* __restrict__ kind_mat, const float* __restrict__ gembeds,
                         float* __restrict__ hs3, float* __restrict__ keyat_all, int t) {
    __shared__ float part[256];
    __shared__ float kl[KK];
    __shared__ float ka[KK];
    __shared__ int anyk;
    const int b = blockIdx.x, tid = threadIdx.x;
    const int k = tid >> 3, u = tid & 7;
    const float* gp = GK + ((size_t)(b * KK + k)) * HH;
    const float* hp = HK + b * HH;
    float p = 0.0f;
    for (int i = u * 64; i < u * 64 + 64; i++) {
        float v = gp[i] + hp[i];
        p += fmaxf(v, 0.0f) * k2w[i];
    }
    part[tid] = p;
    if (tid == 0) anyk = 0;
    __syncthreads();
    if (tid < KK) {
        float s = k2b[0];
        for (int j = 0; j < 8; j++) s += part[tid * 8 + j];
        int notm = (kind_mat[b * KK + tid] != -1);
        if (notm) atomicOr(&anyk, 1);
        kl[tid] = notm ? s : NEGF;
    }
    __syncthreads();
    if (tid == 0) {
        float mx = kl[0];
        for (int j = 1; j < KK; j++) mx = fmaxf(mx, kl[j]);
        float sum = 0.0f;
        for (int j = 0; j < KK; j++) { ka[j] = expf(kl[j] - mx); sum += ka[j]; }
        float inv = anyk ? (1.0f / sum) : 0.0f;
        for (int j = 0; j < KK; j++) ka[j] *= inv;
    }
    __syncthreads();
    if (tid < KK) keyat_all[((size_t)b * TT + t) * KK + tid] = ka[tid];
    for (int h = tid; h < HH; h += 256) {
        float acc = 0.0f;
        for (int j = 0; j < KK; j++)
            acc += ka[j] * gembeds[((size_t)(b * KK + j)) * HH + h];
        hs3[b * H3 + 1024 + h] = acc;
    }
}

// ---------------------------------------------------------------------------
// switch = sigmoid(relu(SWP).s2w + s2b), g = sigmoid(hs3.gw + gb),
// states[b,t] = active ? [hnew, h_att, g_hid] : 0
// ---------------------------------------------------------------------------
__global__ void step_final(const float* __restrict__ SWP, const float* __restrict__ hs3,
                           const float* __restrict__ g_hid,
                           const float* __restrict__ s2w, const float* __restrict__ s2b,
                           const float* __restrict__ gw, const float* __restrict__ gb,
                           const int* __restrict__ r_ulens, int t,
                           float* __restrict__ states, float* __restrict__ sw_all,
                           float* __restrict__ g_all) {
    __shared__ float red[256];
    __shared__ float swv, gv;
    const int b = blockIdx.x, tid = threadIdx.x;
    float p = 0.0f;
    for (int i = tid; i < HH; i += 256) p += fmaxf(SWP[b * HH + i], 0.0f) * s2w[i];
    red[tid] = p;
    __syncthreads();
    for (int off = 128; off > 0; off >>= 1) {
        if (tid < off) red[tid] += red[tid + off];
        __syncthreads();
    }
    if (tid == 0) swv = sigm(red[0] + s2b[0]);
    __syncthreads();
    float q = 0.0f;
    for (int i = tid; i < H3; i += 256) q += hs3[b * H3 + i] * gw[i];
    red[tid] = q;
    __syncthreads();
    for (int off = 128; off > 0; off >>= 1) {
        if (tid < off) red[tid] += red[tid + off];
        __syncthreads();
    }
    if (tid == 0) gv = sigm(red[0] + gb[0]);
    __syncthreads();
    const int active = (t < r_ulens[b]);
    if (tid == 0) {
        sw_all[b * TT + t] = active ? swv : 0.0f;
        g_all[b * TT + t] = gv;
    }
    float* st = states + ((size_t)b * TT + t) * H3;
    for (int c = tid; c < H3; c += 256) {
        float v = (c < 1024) ? hs3[b * H3 + c] : g_hid[b * HH + (c - 1024)];
        st[c] = active ? v : 0.0f;
    }
}

// ---------------------------------------------------------------------------
// Vocab softmax over V per row (3 passes) + scale by (1-ps); then scatter-add
// the copy distribution with atomics.
// ---------------------------------------------------------------------------
__global__ void rowmax_k(const float* __restrict__ X, float* __restrict__ rowmax) {
    __shared__ float red[256];
    const int row = blockIdx.x, tid = threadIdx.x;
    const float* x = X + (size_t)row * VV;
    float m = -3.4e38f;
    for (int i = tid; i < VV; i += 256) m = fmaxf(m, x[i]);
    red[tid] = m;
    __syncthreads();
    for (int off = 128; off > 0; off >>= 1) {
        if (tid < off) red[tid] = fmaxf(red[tid], red[tid + off]);
        __syncthreads();
    }
    if (tid == 0) rowmax[row] = red[0];
}

__global__ void rowexp_k(float* __restrict__ X, const float* __restrict__ rowmax,
                         float* __restrict__ rowsum) {
    __shared__ float red[256];
    const int row = blockIdx.x, tid = threadIdx.x;
    const float mx = rowmax[row];
    float* x = X + (size_t)row * VV;
    float s = 0.0f;
    for (int i = tid; i < VV; i += 256) {
        float e = expf(x[i] - mx);
        x[i] = e;
        s += e;
    }
    red[tid] = s;
    __syncthreads();
    for (int off = 128; off > 0; off >>= 1) {
        if (tid < off) red[tid] += red[tid + off];
        __syncthreads();
    }
    if (tid == 0) rowsum[row] = red[0];
}

__global__ void rowscale_k(float* __restrict__ X, const float* __restrict__ rowsum,
                           const float* __restrict__ sw_all) {
    const int row = blockIdx.x, tid = threadIdx.x;
    const float f = (1.0f - sw_all[row]) / rowsum[row];
    float* x = X + (size_t)row * VV;
    for (int i = tid; i < VV; i += 256) x[i] *= f;
}

__global__ void scatter_copy(float* __restrict__ out,
                             const float* __restrict__ attn_all,
                             const float* __restrict__ keyat_all,
                             const float* __restrict__ sw_all,
                             const float* __restrict__ g_all,
                             const int* __restrict__ c_utts,
                             const int* __restrict__ g_pos) {
    const int row = blockIdx.x;
    const int b = row / TT;
    const int tid = threadIdx.x;
    const float ps = sw_all[row];
    if (ps == 0.0f) return;
    const float g = g_all[row];
    float* o = out + (size_t)row * VV;
    {
        float a = attn_all[(size_t)row * SS + tid];
        float w = ps * (1.0f - g) * a;
        if (w != 0.0f) atomicAdd(o + c_utts[b * SS + tid], w);
    }
    if (tid < KK) {
        float a = keyat_all[(size_t)row * KK + tid];
        float w = ps * g * a;
        if (w != 0.0f) atomicAdd(o + g_pos[b * KK + tid], w);
    }
}

// ---------------------------------------------------------------------------
extern "C" void kernel_launch(void* const* d_in, const int* in_sizes, int n_in,
                              void* d_out, int out_size, void* d_ws, size_t ws_size,
                              hipStream_t stream) {
    (void)in_sizes; (void)n_in; (void)out_size;
    const float* cembeds = (const float*)d_in[0];
    const float* c_uh    = (const float*)d_in[1];
    const float* gembeds = (const float*)d_in[2];
    const float* g_hid   = (const float*)d_in[3];
    const float* emb     = (const float*)d_in[4];
    const float* Wih     = (const float*)d_in[5];
    const float* Whh     = (const float*)d_in[6];
    const float* bih     = (const float*)d_in[7];
    const float* bhh     = (const float*)d_in[8];
    const float* woW     = (const float*)d_in[9];
    const float* wob     = (const float*)d_in[10];
    const float* WhW     = (const float*)d_in[11];
    const float* Whb     = (const float*)d_in[12];
    const float* WsW     = (const float*)d_in[13];
    const float* Wsb     = (const float*)d_in[14];
    const float* Wvw     = (const float*)d_in[15];
    const float* Wvb     = (const float*)d_in[16];
    const float* s1W     = (const float*)d_in[17];
    const float* s1b     = (const float*)d_in[18];
    const float* s2w     = (const float*)d_in[19];
    const float* s2b     = (const float*)d_in[20];
    const float* k1W     = (const float*)d_in[21];
    const float* k1b     = (const float*)d_in[22];
    const float* k2w     = (const float*)d_in[23];
    const float* k2b     = (const float*)d_in[24];
    const float* gw      = (const float*)d_in[25];
    const float* gb      = (const float*)d_in[26];
    const int* r_utts    = (const int*)d_in[27];
    const int* r_ulens   = (const int*)d_in[28];
    const int* c_utts    = (const int*)d_in[29];
    const int* kind_mat  = (const int*)d_in[30];
    const int* g_pos     = (const int*)d_in[31];
    float* out = (float*)d_out;

    float* ws = (float*)d_ws;
    float* dec_in = ws;   ws += (size_t)BB * TT * DIN;
    float* GI = ws;       ws += (size_t)BB * TT * H3;
    float* SX = ws;       ws += (size_t)BB * TT * HH;
    float* HA = ws;       ws += (size_t)BB * SS * HH;
    float* GK = ws;       ws += (size_t)BB * KK * HH;
    float* states = ws;   ws += (size_t)BB * TT * H3;
    float* attn_all = ws; ws += (size_t)BB * TT * SS;
    float* keyat_all = ws; ws += (size_t)BB * TT * KK;
    float* sw_all = ws;   ws += BB * TT;
    float* g_all = ws;    ws += BB * TT;
    float* hbuf0 = ws;    ws += BB * HH;
    float* hbuf1 = ws;    ws += BB * HH;
    float* hs3 = ws;      ws += BB * H3;
    float* STb = ws;      ws += BB * HH;
    float* HKb = ws;      ws += BB * HH;
    float* SWP = ws;      ws += BB * HH;
    float* rmax = ws;     ws += BB * TT;
    float* rsum = ws;     ws += BB * TT;
    unsigned short* wo_bf = (unsigned short*)ws;  // optional bf16 copy of woW
    const size_t base_bytes = (size_t)((float*)wo_bf - (float*)d_ws) * sizeof(float);
    const size_t wo_elems = (size_t)VV * H3;
    const bool use_bf16B = ws_size >= base_bytes + wo_elems * sizeof(unsigned short);

    dim3 thr(256);
    auto nb = [](int waves) { return dim3((unsigned)((waves + 7) / 8)); };

    build_decin<<<(BB * TT * DIN + 255) / 256, thr, 0, stream>>>(cembeds, emb, r_utts, dec_in);
    if (use_bf16B) {
        // one pass; the 98MB bf16 copy stays L2-resident across the 8 strip passes
        cvt_bf16<<<(unsigned)(wo_elems / 8 / 256), thr, 0, stream>>>(woW, wo_bf, wo_elems);
    }
    // GI = dec_in @ Wih^T + bih         (hoisted out of the recurrent loop)
    gemm_bt<1><<<nb((BB * TT / 16) * (H3 / 16)), thr, 0, stream>>>(
        dec_in, DIN, Wih, DIN, bih, nullptr, 0, GI, H3, BB * TT, H3, DIN);
    // SX = dec_in @ s1W[:,3H:]^T        (x-part of the switch MLP)
    gemm_bt<1><<<nb((BB * TT / 16) * (HH / 16)), thr, 0, stream>>>(
        dec_in, DIN, s1W + 3 * HH, SWIN, nullptr, nullptr, 0, SX, HH, BB * TT, HH, DIN);
    // hidden_att = c_uh @ WhW^T + Whb
    gemm_bt<1><<<nb((BB * SS / 16) * (HH / 16)), thr, 0, stream>>>(
        c_uh, HH, WhW, HH, Whb, nullptr, 0, HA, HH, BB * SS, HH, HH);
    // GK = gembeds @ k1W[:,2H:]^T + k1b (key MLP, time-invariant part)
    gemm_bt<1><<<nb((BB * KK / 16) * (HH / 16)), thr, 0, stream>>>(
        gembeds, HH, k1W + 2 * HH, H3, k1b, nullptr, 0, GK, HH, BB * KK, HH, HH);

    for (int t = 0; t < TT; t++) {
        const float* hin = (t == 0) ? cembeds : ((t & 1) ? hbuf0 : hbuf1);
        float* hout = (t & 1) ? hbuf1 : hbuf0;
        gru_step<<<dim3(4), thr, 0, stream>>>(GI + (size_t)t * H3, TT * H3, hin, Whh, bhh,
                                              r_ulens, t, hs3, hout);
        // st = hnew @ WsW^T + Wsb
        gemm_bt<1><<<nb(HH / 16), thr, 0, stream>>>(hs3, H3, WsW, HH, Wsb, nullptr, 0,
                                                    STb, HH, BB, HH, HH);
        attn_step<<<dim3(BB), thr, 0, stream>>>(HA, STb, Wvw, Wvb, c_utts, c_uh, hs3,
                                                attn_all, t);
        // hk = [hnew,h_att] @ k1W[:,:2H]^T
        gemm_bt<1><<<nb(HH / 16), thr, 0, stream>>>(hs3, H3, k1W, H3, nullptr, nullptr, 0,
                                                    HKb, HH, BB, HH, 2 * HH);
        key_step<<<dim3(BB), thr, 0, stream>>>(GK, HKb, k2w, k2b, kind_mat, gembeds, hs3,
                                               keyat_all, t);
        // SWP = [hnew,h_att,h_key] @ s1W[:,:3H]^T + s1b + SX[:,t,:]
        gemm_bt<1><<<nb(HH / 16), thr, 0, stream>>>(hs3, H3, s1W, SWIN, s1b,
                                                    SX + (size_t)t * HH, TT * HH, SWP, HH,
                                                    BB, HH, H3);
        step_final<<<dim3(BB), thr, 0, stream>>>(SWP, hs3, g_hid, s2w, s2b, gw, gb,
                                                 r_ulens, t, states, sw_all, g_all);
    }

    // logits = states @ woW^T + wob -> d_out   (dominant GEMM, MREP=8 strips)
    if (use_bf16B) {
        gemm_btb<8><<<nb((BB * TT / 128) * (VV / 16)), thr, 0, stream>>>(
            states, H3, wo_bf, H3, wob, out, VV, BB * TT, VV, H3);
    } else {
        gemm_bt<8><<<nb((BB * TT / 128) * (VV / 16)), thr, 0, stream>>>(
            states, H3, woW, H3, wob, nullptr, 0, out, VV, BB * TT, VV, H3);
    }
    rowmax_k<<<dim3(BB * TT), thr, 0, stream>>>(out, rmax);
    rowexp_k<<<dim3(BB * TT), thr, 0, stream>>>(out, rmax, rsum);
    rowscale_k<<<dim3(BB * TT), thr, 0, stream>>>(out, rsum, sw_all);
    scatter_copy<<<dim3(BB * TT), thr, 0, stream>>>(out, attn_all, keyat_all, sw_all,
                                                    g_all, c_utts, g_pos);
}
